// MultiHeadLatentAttention_34720515621032
// MI455X (gfx1250) — matrix operand order repaired and checked by reading, not compiled, not run on hardware
//
#include <hip/hip_runtime.h>
#include <hip/hip_bf16.h>

// ---------------------------------------------------------------------------
// MLA (multi-head latent attention) for MI455X / gfx1250, wave32 + WMMA.
// ---------------------------------------------------------------------------

#define T_      2048
#define D_      2560
#define NH_     20
#define HD_     128
#define R_      64
#define CQ_     768
#define CKV_    512
#define WINDOW_ 4096
#define NHD_    2560   // NH*HD
#define NR_     1280   // NH*R
#define DK_     192    // HD+R

typedef __attribute__((ext_vector_type(16))) _Float16 v16h;
typedef __attribute__((ext_vector_type(8)))  _Float16 v8h;
typedef __attribute__((ext_vector_type(8)))  float    v8f;

union Frag16 { v16h v; v8h h[2]; };

// ---------------------------- elementwise ----------------------------------

__global__ void k_to_f16(const float* __restrict__ in, _Float16* __restrict__ out, int n) {
  int i = blockIdx.x * blockDim.x + threadIdx.x;
  if (i < n) out[i] = (_Float16)in[i];
}

// in: K x N (f32, row-major)  ->  out: N x K (f16, row-major)
__global__ void k_transpose_f16(const float* __restrict__ in, _Float16* __restrict__ out,
                                int K, int N) {
  int i = blockIdx.x * blockDim.x + threadIdx.x;
  if (i >= K * N) return;
  int k = i / N;
  int n = i - k * N;
  out[(size_t)n * K + k] = (_Float16)in[i];
}

// wave-per-row RMSNorm, f32 in -> f16 out
__global__ void k_rmsnorm_f16(const float* __restrict__ in, const float* __restrict__ scale,
                              _Float16* __restrict__ out, int rows, int C) {
  int row  = blockIdx.x * (blockDim.x >> 5) + (threadIdx.x >> 5);
  int lane = threadIdx.x & 31;
  if (row >= rows) return;
  const float* r = in + (size_t)row * C;
  float s = 0.f;
  for (int c = lane; c < C; c += 32) { float v = r[c]; s += v * v; }
  s += __shfl_xor(s, 1, 32);
  s += __shfl_xor(s, 2, 32);
  s += __shfl_xor(s, 4, 32);
  s += __shfl_xor(s, 8, 32);
  s += __shfl_xor(s, 16, 32);
  float rms = rsqrtf(s / (float)C + 1e-6f);
  _Float16* o = out + (size_t)row * C;
  for (int c = lane; c < C; c += 32) o[c] = (_Float16)(r[c] * rms * scale[c]);
}

// alpha[t] = sigmoid(x[t,:] . gate_w + gate_b)
__global__ void k_gate(const float* __restrict__ x, const float* __restrict__ gw,
                       const float* __restrict__ gb, float* __restrict__ alpha, int rows) {
  int row  = blockIdx.x * (blockDim.x >> 5) + (threadIdx.x >> 5);
  int lane = threadIdx.x & 31;
  if (row >= rows) return;
  const float* r = x + (size_t)row * D_;
  float s = 0.f;
  for (int c = lane; c < D_; c += 32) s += r[c] * gw[c];
  s += __shfl_xor(s, 1, 32);
  s += __shfl_xor(s, 2, 32);
  s += __shfl_xor(s, 4, 32);
  s += __shfl_xor(s, 8, 32);
  s += __shfl_xor(s, 16, 32);
  if (lane == 0) alpha[row] = 1.f / (1.f + __expf(-(s + gb[0])));
}

// Q/K assembly: out (T, NH, 192) f16; [0,128) from up-proj, [128,192) roped.
__global__ void k_assemble_qk(const float* __restrict__ c32, const float* __restrict__ r32,
                              _Float16* __restrict__ out, int n) {
  int idx = blockIdx.x * blockDim.x + threadIdx.x;
  if (idx >= n) return;
  int c  = idx % DK_;
  int th = idx / DK_;
  int h  = th % NH_;
  int t  = th / NH_;
  float val;
  if (c < HD_) {
    val = c32[(size_t)t * NHD_ + h * HD_ + c];
  } else {
    int cr = c - HD_;            // 0..63
    int i  = cr & 31;
    float xr = r32[(size_t)t * NR_ + h * R_ + i];
    float xi = r32[(size_t)t * NR_ + h * R_ + 32 + i];
    float ang = (float)t * powf(500000.0f, -(float)i * (1.0f / 32.0f));
    float cs = cosf(ang);
    float sn = sinf(ang);
    val = (cr < 32) ? (xr * cs - xi * sn) : (xr * sn + xi * cs);
  }
  out[idx] = (_Float16)val;
}

// Vt[(h*HD+d)*T + t] = v32[t, h*HD+d]
__global__ void k_build_vt(const float* __restrict__ v32, _Float16* __restrict__ Vt, int n) {
  int idx = blockIdx.x * blockDim.x + threadIdx.x;
  if (idx >= n) return;
  int t  = idx % T_;
  int hd = idx / T_;
  Vt[idx] = (_Float16)v32[(size_t)t * NHD_ + hd];
}

// ------------------------------- GEMM --------------------------------------
// C[M,N](f32) = A[M,K](f16 row-major) x Bt[N,K](f16 row-major, i.e. B^T).
// Wave computes 32x32; 4 waves -> 64x64 block tile. M%64==0, N%64==0, K%32==0.
__global__ __launch_bounds__(128) void k_gemm(const _Float16* __restrict__ A,
                                              const _Float16* __restrict__ Bt,
                                              float* __restrict__ C,
                                              int M, int N, int K) {
  (void)M;
  const int lane = threadIdx.x & 31;
  const int wave = threadIdx.x >> 5;
  const int l15  = lane & 15;
  const bool hi  = lane >= 16;
  const int waveM = blockIdx.y * 64 + (wave >> 1) * 32;
  const int waveN = blockIdx.x * 64 + (wave & 1) * 32;
  const int ka  = hi ? 16 : 0;   // A-frag K offset, contiguous k = 16h + s
  const int kbo = hi ? 16 : 0;   // B-frag K offset, contiguous k = 16h + s

  const _Float16* pa0 = A  + (size_t)(waveM + l15) * K + ka;
  const _Float16* pa1 = pa0 + (size_t)16 * K;
  const _Float16* pb0 = Bt + (size_t)(waveN + l15) * K + kbo;
  const _Float16* pb1 = pb0 + (size_t)16 * K;

  v8f c00 = {}, c01 = {}, c10 = {}, c11 = {};
  for (int k0 = 0; k0 < K; k0 += 32) {
    Frag16 a0, a1;
    a0.h[0] = *(const v8h*)(pa0 + k0);
    a0.h[1] = *(const v8h*)(pa0 + k0 + 8);
    a1.h[0] = *(const v8h*)(pa1 + k0);
    a1.h[1] = *(const v8h*)(pa1 + k0 + 8);
    v16h b0 = *(const v16h*)(pb0 + k0);
    v16h b1 = *(const v16h*)(pb1 + k0);
    __builtin_prefetch(pa0 + k0 + 128, 0, 1);
    __builtin_prefetch(pb0 + k0 + 128, 0, 1);
    c00 = __builtin_amdgcn_wmma_f32_16x16x32_f16(false, a0.v, false, b0, (short)0, c00, false, false);
    c01 = __builtin_amdgcn_wmma_f32_16x16x32_f16(false, a0.v, false, b1, (short)0, c01, false, false);
    c10 = __builtin_amdgcn_wmma_f32_16x16x32_f16(false, a1.v, false, b0, (short)0, c10, false, false);
    c11 = __builtin_amdgcn_wmma_f32_16x16x32_f16(false, a1.v, false, b1, (short)0, c11, false, false);
  }
  const int mb = waveM + (hi ? 8 : 0);
  const int nc = waveN + l15;
#pragma unroll
  for (int i = 0; i < 8; ++i) {
    C[(size_t)(mb + i) * N + nc]           = c00[i];
    C[(size_t)(mb + i) * N + nc + 16]      = c01[i];
    C[(size_t)(mb + 16 + i) * N + nc]      = c10[i];
    C[(size_t)(mb + 16 + i) * N + nc + 16] = c11[i];
  }
}

// ----------------------------- attention -----------------------------------

__device__ __forceinline__ void online_update(const float* sc0, const float* sc1,
                                              float* m, float* l, v8f* o,
                                              _Float16* Pt, int l15, int rbase) {
#pragma unroll
  for (int i = 0; i < 8; ++i) {
    float rm = fmaxf(sc0[i], sc1[i]);
    rm = fmaxf(rm, __shfl_xor(rm, 1, 32));
    rm = fmaxf(rm, __shfl_xor(rm, 2, 32));
    rm = fmaxf(rm, __shfl_xor(rm, 4, 32));
    rm = fmaxf(rm, __shfl_xor(rm, 8, 32));
    float mn   = fmaxf(m[i], rm);
    float corr = __expf(m[i] - mn);
    float p0   = __expf(sc0[i] - mn);
    float p1   = __expf(sc1[i] - mn);
    float rs   = p0 + p1;
    rs += __shfl_xor(rs, 1, 32);
    rs += __shfl_xor(rs, 2, 32);
    rs += __shfl_xor(rs, 4, 32);
    rs += __shfl_xor(rs, 8, 32);
    l[i] = l[i] * corr + rs;
    m[i] = mn;
#pragma unroll
    for (int j = 0; j < 8; ++j) o[j][i] *= corr;
    Pt[(rbase + i) * 32 + l15]      = (_Float16)p0;
    Pt[(rbase + i) * 32 + 16 + l15] = (_Float16)p1;
  }
}

// One wave per (head, 16-row Q tile). Streams 32-key blocks; shared raw scores,
// two online softmaxes (causal "ar" + bidirectional "bi"), gate-blended output.
__global__ __launch_bounds__(32) void k_attn(const _Float16* __restrict__ Q,
                                             const _Float16* __restrict__ Km,
                                             const _Float16* __restrict__ Vt,
                                             const float* __restrict__ alpha,
                                             _Float16* __restrict__ merged) {
  const int head  = blockIdx.x;
  const int qb    = blockIdx.y * 16;
  const int lane  = threadIdx.x;
  const int l15   = lane & 15;
  const bool hi   = lane >= 16;
  const int rbase = hi ? 8 : 0;
  const int ka    = hi ? 16 : 0;
  const int kbo   = hi ? 16 : 0;

  __shared__ _Float16 PtA[16 * 32];
  __shared__ _Float16 PtB[16 * 32];

  // cache the 16x192 Q tile as 6 A-fragments
  Frag16 qf[6];
  {
    const _Float16* qp = Q + ((size_t)(qb + l15) * NH_ + head) * DK_ + ka;
#pragma unroll
    for (int kc = 0; kc < 6; ++kc) {
      qf[kc].h[0] = *(const v8h*)(qp + kc * 32);
      qf[kc].h[1] = *(const v8h*)(qp + kc * 32 + 8);
    }
  }

  float m_ar[8], l_ar[8], m_bi[8], l_bi[8];
  v8f o_ar[8], o_bi[8];
  v8f vzero = {};
#pragma unroll
  for (int i = 0; i < 8; ++i) {
    m_ar[i] = -1e30f; m_bi[i] = -1e30f;
    l_ar[i] = 0.f;    l_bi[i] = 0.f;
    o_ar[i] = vzero;  o_bi[i] = vzero;
  }
  const float sAR = 0.08838834764831845f;  // HD^-0.5
  const float sBI = 0.07216878364870323f;  // (HD+R)^-0.5

  for (int kb = 0; kb < T_; kb += 32) {
    // raw scores: two 16x16 tiles over the 192-dim contraction
    v8f s0 = vzero, s1 = vzero;
#pragma unroll
    for (int kc = 0; kc < 6; ++kc) {
      const _Float16* kp0 = Km + ((size_t)(kb + l15) * NH_ + head) * DK_ + kc * 32 + kbo;
      const _Float16* kp1 = Km + ((size_t)(kb + 16 + l15) * NH_ + head) * DK_ + kc * 32 + kbo;
      v16h b0 = *(const v16h*)kp0;
      v16h b1 = *(const v16h*)kp1;
      s0 = __builtin_amdgcn_wmma_f32_16x16x32_f16(false, qf[kc].v, false, b0, (short)0, s0, false, false);
      s1 = __builtin_amdgcn_wmma_f32_16x16x32_f16(false, qf[kc].v, false, b1, (short)0, s1, false, false);
    }

    const bool doAR = (kb <= qb);  // key blocks fully above the Q tile contribute 0 to "ar"
    if (doAR) {
      float sc0[8], sc1[8];
#pragma unroll
      for (int i = 0; i < 8; ++i) {
        int ig = qb + rbase + i;
        int j0 = kb + l15;
        int j1 = j0 + 16;
        sc0[i] = (j0 <= ig && (ig - j0) < WINDOW_) ? s0[i] * sAR : -1e30f;
        sc1[i] = (j1 <= ig && (ig - j1) < WINDOW_) ? s1[i] * sAR : -1e30f;
      }
      online_update(sc0, sc1, m_ar, l_ar, o_ar, PtA, l15, rbase);
    }
    {
      float sc0[8], sc1[8];
#pragma unroll
      for (int i = 0; i < 8; ++i) { sc0[i] = s0[i] * sBI; sc1[i] = s1[i] * sBI; }
      online_update(sc0, sc1, m_bi, l_bi, o_bi, PtB, l15, rbase);
    }

    // re-shape P tiles (C-layout -> A-layout) through wave-private LDS
    Frag16 pA, pB;
    if (doAR) {
      pA.h[0] = *(const v8h*)&PtA[l15 * 32 + ka];
      pA.h[1] = *(const v8h*)&PtA[l15 * 32 + ka + 8];
    }
    pB.h[0] = *(const v8h*)&PtB[l15 * 32 + ka];
    pB.h[1] = *(const v8h*)&PtB[l15 * 32 + ka + 8];

    // P x V : 8 output d-tiles of 16 columns, V fragments shared by both branches
#pragma unroll
    for (int j = 0; j < 8; ++j) {
      const _Float16* vp = Vt + (size_t)(head * HD_ + j * 16 + l15) * T_ + kb + kbo;
      v16h bv = *(const v16h*)vp;
      if (doAR)
        o_ar[j] = __builtin_amdgcn_wmma_f32_16x16x32_f16(false, pA.v, false, bv, (short)0, o_ar[j], false, false);
      o_bi[j] = __builtin_amdgcn_wmma_f32_16x16x32_f16(false, pB.v, false, bv, (short)0, o_bi[j], false, false);
    }
  }

  // gate-blend the two normalized outputs and store f16 for the final GEMM
#pragma unroll
  for (int i = 0; i < 8; ++i) {
    int row   = qb + rbase + i;
    float a   = alpha[row];
    float iar = 1.f / l_ar[i];
    float ibi = 1.f / l_bi[i];
    _Float16* mp = merged + (size_t)row * NHD_ + head * HD_ + l15;
#pragma unroll
    for (int j = 0; j < 8; ++j)
      mp[j * 16] = (_Float16)(a * o_bi[j][i] * ibi + (1.f - a) * o_ar[j][i] * iar);
  }
}

// ------------------------------- launch ------------------------------------

extern "C" void kernel_launch(void* const* d_in, const int* in_sizes, int n_in,
                              void* d_out, int out_size, void* d_ws, size_t ws_size,
                              hipStream_t stream) {
  (void)in_sizes; (void)n_in; (void)out_size; (void)ws_size;
  const float* x    = (const float*)d_in[0];
  const float* qdw  = (const float*)d_in[1];
  const float* qns  = (const float*)d_in[2];
  const float* quw  = (const float*)d_in[3];
  const float* qrw  = (const float*)d_in[4];
  const float* kvdw = (const float*)d_in[5];
  const float* kvns = (const float*)d_in[6];
  const float* kuw  = (const float*)d_in[7];
  const float* vuw  = (const float*)d_in[8];
  const float* krw  = (const float*)d_in[9];
  const float* outw = (const float*)d_in[10];
  const float* gw   = (const float*)d_in[11];
  const float* gb   = (const float*)d_in[12];
  float* out = (float*)d_out;

  size_t off = 0;
  auto carve = [&](size_t bytes) -> void* {
    void* p = (char*)d_ws + off;
    off += (bytes + 255) & ~(size_t)255;
    return p;
  };
  _Float16* xh      = (_Float16*)carve((size_t)T_ * D_ * 2);
  _Float16* wqd     = (_Float16*)carve((size_t)CQ_ * D_ * 2);
  _Float16* wqu     = (_Float16*)carve((size_t)NHD_ * CQ_ * 2);
  _Float16* wqr     = (_Float16*)carve((size_t)NR_ * CQ_ * 2);
  _Float16* wkvd    = (_Float16*)carve((size_t)CKV_ * D_ * 2);
  _Float16* wku     = (_Float16*)carve((size_t)NHD_ * CKV_ * 2);
  _Float16* wvu     = (_Float16*)carve((size_t)NHD_ * CKV_ * 2);
  _Float16* wkr     = (_Float16*)carve((size_t)NR_ * D_ * 2);
  _Float16* wout    = (_Float16*)carve((size_t)D_ * NHD_ * 2);
  float*    qlat32  = (float*)carve((size_t)T_ * CQ_ * 4);
  _Float16* qlat16  = (_Float16*)carve((size_t)T_ * CQ_ * 2);
  float*    kvlat32 = (float*)carve((size_t)T_ * CKV_ * 4);
  _Float16* kvlat16 = (_Float16*)carve((size_t)T_ * CKV_ * 2);
  float*    qc32    = (float*)carve((size_t)T_ * NHD_ * 4);
  float*    qr32    = (float*)carve((size_t)T_ * NR_ * 4);
  float*    kc32    = (float*)carve((size_t)T_ * NHD_ * 4);
  float*    v32     = (float*)carve((size_t)T_ * NHD_ * 4);
  float*    kr32    = (float*)carve((size_t)T_ * NR_ * 4);
  _Float16* Qh      = (_Float16*)carve((size_t)T_ * NH_ * DK_ * 2);
  _Float16* Kh      = (_Float16*)carve((size_t)T_ * NH_ * DK_ * 2);
  _Float16* Vt      = (_Float16*)carve((size_t)NH_ * HD_ * T_ * 2);
  _Float16* merged  = (_Float16*)carve((size_t)T_ * NHD_ * 2);
  float*    alpha   = (float*)carve((size_t)T_ * 4);

  auto cdiv = [](int a, int b) { return (a + b - 1) / b; };

  // precision conversion + weight transposition (B^T layout for WMMA B-frags)
  k_to_f16<<<cdiv(T_ * D_, 256), 256, 0, stream>>>(x, xh, T_ * D_);
  k_transpose_f16<<<cdiv(D_ * CQ_, 256), 256, 0, stream>>>(qdw, wqd, D_, CQ_);
  k_transpose_f16<<<cdiv(CQ_ * NHD_, 256), 256, 0, stream>>>(quw, wqu, CQ_, NHD_);
  k_transpose_f16<<<cdiv(CQ_ * NR_, 256), 256, 0, stream>>>(qrw, wqr, CQ_, NR_);
  k_transpose_f16<<<cdiv(D_ * CKV_, 256), 256, 0, stream>>>(kvdw, wkvd, D_, CKV_);
  k_transpose_f16<<<cdiv(CKV_ * NHD_, 256), 256, 0, stream>>>(kuw, wku, CKV_, NHD_);
  k_transpose_f16<<<cdiv(CKV_ * NHD_, 256), 256, 0, stream>>>(vuw, wvu, CKV_, NHD_);
  k_transpose_f16<<<cdiv(D_ * NR_, 256), 256, 0, stream>>>(krw, wkr, D_, NR_);
  k_transpose_f16<<<cdiv(NHD_ * D_, 256), 256, 0, stream>>>(outw, wout, NHD_, D_);
  k_gate<<<cdiv(T_, 8), 256, 0, stream>>>(x, gw, gb, alpha, T_);

  dim3 blk(128);
  // q path
  k_gemm<<<dim3(CQ_ / 64, T_ / 64), blk, 0, stream>>>(xh, wqd, qlat32, T_, CQ_, D_);
  k_rmsnorm_f16<<<cdiv(T_, 8), 256, 0, stream>>>(qlat32, qns, qlat16, T_, CQ_);
  k_gemm<<<dim3(NHD_ / 64, T_ / 64), blk, 0, stream>>>(qlat16, wqu, qc32, T_, NHD_, CQ_);
  k_gemm<<<dim3(NR_ / 64, T_ / 64), blk, 0, stream>>>(qlat16, wqr, qr32, T_, NR_, CQ_);
  // kv path
  k_gemm<<<dim3(CKV_ / 64, T_ / 64), blk, 0, stream>>>(xh, wkvd, kvlat32, T_, CKV_, D_);
  k_rmsnorm_f16<<<cdiv(T_, 8), 256, 0, stream>>>(kvlat32, kvns, kvlat16, T_, CKV_);
  k_gemm<<<dim3(NHD_ / 64, T_ / 64), blk, 0, stream>>>(kvlat16, wku, kc32, T_, NHD_, CKV_);
  k_gemm<<<dim3(NHD_ / 64, T_ / 64), blk, 0, stream>>>(kvlat16, wvu, v32, T_, NHD_, CKV_);
  k_gemm<<<dim3(NR_ / 64, T_ / 64), blk, 0, stream>>>(xh, wkr, kr32, T_, NR_, D_);

  // rope + pack Q/K, transpose V
  int nq = T_ * NH_ * DK_;
  k_assemble_qk<<<cdiv(nq, 256), 256, 0, stream>>>(qc32, qr32, Qh, nq);
  k_assemble_qk<<<cdiv(nq, 256), 256, 0, stream>>>(kc32, kr32, Kh, nq);
  int nv = NH_ * HD_ * T_;
  k_build_vt<<<cdiv(nv, 256), 256, 0, stream>>>(v32, Vt, nv);

  // dual-softmax flash attention + gate blend
  k_attn<<<dim3(NH_, T_ / 16), dim3(32), 0, stream>>>(Qh, Kh, Vt, alpha, merged);

  // output projection straight into d_out (f32)
  k_gemm<<<dim3(D_ / 64, T_ / 64), blk, 0, stream>>>(merged, wout, out, T_, D_, D_);
}
